// Attention_45183055954094
// MI455X (gfx1250) — matrix operand order repaired and checked by reading, not compiled, not run on hardware
//
#include <hip/hip_runtime.h>

// ---------------------------------------------------------------------------
// Shapes: b*n = 64 batches, S = 1024 tokens, C = 128 channels, D = 32 head dim
// ---------------------------------------------------------------------------
#define NBATCH 64
#define SLEN   1024
#define CDIM   128
#define DHEAD  32
#define SPITCH 1028      // score LDS row pitch (pitch%8!=0 -> no 8-row bank hit)
#define NWAVE  4         // waves per attention block

typedef __attribute__((ext_vector_type(16))) __bf16 v16bf;
typedef __attribute__((ext_vector_type(8)))  float  v8f;

union FragB { v16bf v; unsigned int u[8]; };

// pack two floats into a dword of bf16 (round-to-nearest-even)
__device__ __forceinline__ unsigned int pk_bf2(float a, float b) {
  unsigned int ua = __float_as_uint(a);
  unsigned int ub = __float_as_uint(b);
  ua = (ua + 0x7FFFu + ((ua >> 16) & 1u)) >> 16;
  ub = (ub + 0x7FFFu + ((ub >> 16) & 1u)) & 0xFFFF0000u;
  return (ua & 0xFFFFu) | ub;
}

__device__ __forceinline__ v8f wmma_bf16(const FragB& a, const FragB& b, v8f c) {
  return __builtin_amdgcn_wmma_f32_16x16x32_bf16(
      /*neg_a=*/false, a.v, /*neg_b=*/false, b.v,
      /*c_mod=*/(short)0, c, /*reuse_a=*/false, /*reuse_b=*/false);
}

// A-fragment (16-bit, 16x32) K index for vgpr j, lane-half hi (ISA 7.12.2)
__device__ __forceinline__ int kpatA(int j, int hi) {
  return (j < 4) ? (2 * j + 8 * hi) : (16 + 2 * (j - 4) + 8 * hi);
}

// ---------------------------------------------------------------------------
// Kernel 0: V = (f2 + t_pos2) packed to channel-major bf16, pairs over tokens
// (== WMMA B-fragment dword layout).  Fully coalesced.
// ---------------------------------------------------------------------------
__global__ __launch_bounds__(256) void pack_v_kernel(
    const float* __restrict__ f2, const float* __restrict__ tp2,
    unsigned int* __restrict__ Vbf) {
  const int i   = blockIdx.x * 256 + threadIdx.x;  // dword index (2 tokens)
  const int row = i >> 9;                          // (b*128 + c), 512 dw/row
  const float tp = tp2[row & 127];
  const float2 t = *(const float2*)(f2 + row * SLEN + (i & 511) * 2);
  Vbf[i] = pk_bf2(t.x + tp, t.y + tp);
}

// ---------------------------------------------------------------------------
// Kernel 1: Out^T(32 x S) = W(32x128) x (f_l + t_pos)^T(128 x S)
// channel-major input == natural WMMA B layout.  bias + l2norm + (Q: *30),
// stored token-major bf16: Qw/Kw[b][s][d].
// ---------------------------------------------------------------------------
__global__ __launch_bounds__(256) void proj_qk_kernel(
    const float* __restrict__ f1, const float* __restrict__ f2,
    const float* __restrict__ tp1, const float* __restrict__ tp2,
    const float* __restrict__ W, const float* __restrict__ bias,
    unsigned short* __restrict__ Qw, unsigned short* __restrict__ Kw,
    float* __restrict__ ssq) {
  if (blockIdx.x == 0 && threadIdx.x < NBATCH) ssq[threadIdx.x] = 0.f;

  const int b    = blockIdx.x >> 3;
  const int sg   = blockIdx.x & 7;
  const int wave = threadIdx.x >> 5;
  const int lane = threadIdx.x & 31;
  const int s0   = sg * 128 + wave * 16;
  const int n    = lane & 15;   // A-role: row m (=d), B-role: col n (=token)
  const int hi   = lane >> 4;

  // W as A-fragments: 2 d-tiles x 4 c-chunks (pairs over c contiguous)
  FragB wf[2][4];
#pragma unroll
  for (int dt = 0; dt < 2; ++dt) {
    const float* wr = W + (dt * 16 + n) * CDIM;
#pragma unroll
    for (int cc = 0; cc < 4; ++cc)
#pragma unroll
      for (int j = 0; j < 8; ++j) {
        int c = cc * 32 + kpatA(j, hi);
        wf[dt][cc].u[j] = pk_bf2(wr[c], wr[c + 1]);
      }
  }

  const int bo = b * CDIM * SLEN;
#pragma unroll
  for (int src = 0; src < 2; ++src) {
    const float* fs = src ? f2 : f1;
    const float* tp = src ? tp2 : tp1;
    unsigned short* dst = src ? Kw : Qw;

    v8f acc0 = {}, acc1 = {};
    for (int cc = 0; cc < 4; ++cc) {
      FragB bf;
#pragma unroll
      for (int j = 0; j < 8; ++j) {   // B pairs over K=c: two strided loads
        int c = cc * 32 + kpatA(j, hi);
        float x0 = fs[bo + c * SLEN + s0 + n] + tp[c];
        float x1 = fs[bo + (c + 1) * SLEN + s0 + n] + tp[c + 1];
        bf.u[j] = pk_bf2(x0, x1);
      }
      acc0 = wmma_bf16(wf[0][cc], bf, acc0);
      acc1 = wmma_bf16(wf[1][cc], bf, acc1);
    }

    // bias + L2 normalize across d (16 vals here + 16 in lane^16)
    float v0[8], v1[8], ss = 0.f;
#pragma unroll
    for (int r = 0; r < 8; ++r) {
      v0[r] = acc0[r] + bias[r + 8 * hi];
      v1[r] = acc1[r] + bias[16 + r + 8 * hi];
      ss += v0[r] * v0[r] + v1[r] * v1[r];
    }
    ss += __shfl_xor(ss, 16, 32);
    float inv = 1.f / fmaxf(sqrtf(ss), 1e-12f);
    float sc = (src == 0) ? 30.f * inv : inv;   // fold TEMP into Q

    unsigned int* out = (unsigned int*)(dst + (b * SLEN + s0 + n) * DHEAD);
#pragma unroll
    for (int r = 0; r < 8; r += 2) {
      out[(8 * hi + r) >> 1]      = pk_bf2(v0[r] * sc, v0[r + 1] * sc);
      out[(16 + 8 * hi + r) >> 1] = pk_bf2(v1[r] * sc, v1[r + 1] * sc);
    }
  }
}

// ---------------------------------------------------------------------------
// Kernel 2: per-wave 16 queries; scores (f32, LDS-resident) -> softmax ->
// P x V with V pre-packed bf16 (zero-VALU B fragments).
// O[b][s][c] f32 with 1/rowsum folded in.
// ---------------------------------------------------------------------------
__global__ __launch_bounds__(128) void attn_kernel(
    const unsigned short* __restrict__ Qw, const unsigned short* __restrict__ Kw,
    const unsigned int* __restrict__ Vbf, float* __restrict__ O) {
  extern __shared__ float smem[];
  const int b    = blockIdx.x >> 4;
  const int qg   = blockIdx.x & 15;
  const int wave = threadIdx.x >> 5;
  const int lane = threadIdx.x & 31;
  const int q0   = qg * 64 + wave * 16;
  const int n    = lane & 15;
  const int hi   = lane >> 4;

  float* S    = smem + wave * 16 * SPITCH;
  float* invs = smem + NWAVE * 16 * SPITCH + wave * 16;

  // Q A-fragment: rows = queries, K = d (pairs over d contiguous, dword loads)
  FragB qa;
  {
    const unsigned int* qr =
        (const unsigned int*)(Qw + (b * SLEN + q0 + n) * DHEAD);
#pragma unroll
    for (int j = 0; j < 8; ++j) qa.u[j] = qr[kpatA(j, hi) >> 1];
  }

  // ---- scores: 64 x (16q x 16k) WMMA tiles ----
  for (int kt = 0; kt < 64; ++kt) {
    FragB kb;
    const unsigned int* kr =
        (const unsigned int*)(Kw + (b * SLEN + kt * 16 + n) * DHEAD);
#pragma unroll
    for (int j = 0; j < 8; ++j) kb.u[j] = kr[kpatA(j, hi) >> 1];
    v8f acc = {};
    acc = wmma_bf16(qa, kb, acc);
#pragma unroll
    for (int r = 0; r < 8; ++r)
      S[(r + 8 * hi) * SPITCH + kt * 16 + n] = acc[r];
  }
  __syncthreads();   // cross-lane LDS dependence (rows written by other lanes)

  // ---- softmax: 2 lanes per row, 512 keys each ----
  {
    float* Sr = S + n * SPITCH + hi * 512;
    float mx = -3.0e38f;
    for (int i = 0; i < 512; i += 4) {
      float4 t = *(const float4*)(Sr + i);
      mx = fmaxf(mx, fmaxf(fmaxf(t.x, t.y), fmaxf(t.z, t.w)));
    }
    mx = fmaxf(mx, __shfl_xor(mx, 16, 32));
    float sum = 0.f;
    for (int i = 0; i < 512; i += 4) {
      float4 t = *(float4*)(Sr + i);
      t.x = exp2f((t.x - mx) * 1.44269504f);
      t.y = exp2f((t.y - mx) * 1.44269504f);
      t.z = exp2f((t.z - mx) * 1.44269504f);
      t.w = exp2f((t.w - mx) * 1.44269504f);
      sum += t.x + t.y + t.z + t.w;
      *(float4*)(Sr + i) = t;
    }
    sum += __shfl_xor(sum, 16, 32);
    invs[n] = 1.f / sum;   // both halves store the same value
  }
  __syncthreads();

  float osc[8];
#pragma unroll
  for (int r = 0; r < 8; ++r) osc[r] = invs[r + 8 * hi];

  // ---- P x V: 32 key-chunks x 8 channel-chunks ----
  v8f accO[8];
#pragma unroll
  for (int cc = 0; cc < 8; ++cc) { v8f z = {}; accO[cc] = z; }

  // V B-fragment base: dword index into channel-major packed V
  const unsigned int* Vb = Vbf + (b * CDIM) * (SLEN / 2);
  const float* Sa = S + n * SPITCH;

  for (int kc = 0; kc < 32; ++kc) {
    FragB pa;   // P as A: rows = queries, K = keys (f32 pairs from LDS)
#pragma unroll
    for (int j = 0; j < 8; ++j) {
      int k = kc * 32 + 16 * hi + 2 * j;
      float2 t = *(const float2*)(Sa + k);
      pa.u[j] = pk_bf2(t.x, t.y);
    }
#pragma unroll
    for (int cc = 0; cc < 8; ++cc) {
      FragB vb;   // V as B: pre-packed bf16 pairs over keys -> pure loads
      const unsigned int* vr =
          Vb + (cc * 16 + n) * (SLEN / 2) + kc * 16 + 8 * hi;
#pragma unroll
      for (int j = 0; j < 8; ++j) vb.u[j] = vr[j];
      accO[cc] = wmma_bf16(pa, vb, accO[cc]);
    }
  }

  float* Ob = O + b * SLEN * CDIM;
#pragma unroll
  for (int cc = 0; cc < 8; ++cc)
#pragma unroll
    for (int r = 0; r < 8; ++r)
      Ob[(q0 + r + 8 * hi) * CDIM + cc * 16 + n] = accO[cc][r] * osc[r];
}

// ---------------------------------------------------------------------------
// Kernel 3: transpose O back to channel-major, add residual f_l2 (fp32 path),
// write un-normalized output, accumulate per-batch sum of squares.
// ---------------------------------------------------------------------------
__global__ __launch_bounds__(256) void resid_kernel(
    const float* __restrict__ O, const float* __restrict__ f2,
    const float* __restrict__ tp2, float* __restrict__ out,
    float* __restrict__ ssq) {
  __shared__ float tile[32][33];
  __shared__ float wsum[8];
  const int b   = blockIdx.x >> 7;
  const int rem = blockIdx.x & 127;
  const int c0  = (rem >> 5) * 32;
  const int s0  = (rem & 31) * 32;
  const int tx  = threadIdx.x & 31;
  const int ty  = threadIdx.x >> 5;
  const int bo  = b * CDIM * SLEN;

#pragma unroll
  for (int k = 0; k < 4; ++k) {
    int sl = ty + k * 8;
    tile[sl][tx] = O[bo + (s0 + sl) * CDIM + c0 + tx];   // coalesced over c
  }
  __syncthreads();

  float acc = 0.f;
#pragma unroll
  for (int k = 0; k < 4; ++k) {
    int cl = ty + k * 8;
    int c = c0 + cl;
    float v = tile[tx][cl] + f2[bo + c * SLEN + s0 + tx] + tp2[c];
    acc += v * v;
    out[bo + c * SLEN + s0 + tx] = v;                    // coalesced over s
  }
#pragma unroll
  for (int off = 16; off > 0; off >>= 1) acc += __shfl_xor(acc, off, 32);
  if (tx == 0) wsum[ty] = acc;
  __syncthreads();
  if (threadIdx.x == 0) {
    float t = 0.f;
#pragma unroll
    for (int i = 0; i < 8; ++i) t += wsum[i];
    atomicAdd(ssq + b, t);
  }
}

// ---------------------------------------------------------------------------
// Kernel 4: InstanceL2Norm scale, float4 vectorized.
// ---------------------------------------------------------------------------
__global__ __launch_bounds__(256) void scale_kernel(
    float* __restrict__ out, const float* __restrict__ ssq) {
  const int i4 = blockIdx.x * 256 + threadIdx.x;   // float4 index
  const int b  = i4 >> 15;                         // 131072/4 per batch
  const float f =
      0.022097086912079608f * sqrtf(131072.f / (ssq[b] + 1e-5f));
  float4* p = (float4*)out + i4;
  float4 v = *p;
  v.x *= f; v.y *= f; v.z *= f; v.w *= f;
  *p = v;
}

// ---------------------------------------------------------------------------
extern "C" void kernel_launch(void* const* d_in, const int* in_sizes, int n_in,
                              void* d_out, int out_size, void* d_ws,
                              size_t ws_size, hipStream_t stream) {
  const float* f1   = (const float*)d_in[0];
  const float* f2   = (const float*)d_in[1];
  const float* tp1  = (const float*)d_in[2];
  const float* tp2  = (const float*)d_in[3];
  const float* W    = (const float*)d_in[4];
  const float* bias = (const float*)d_in[5];
  float* out = (float*)d_out;

  // workspace: Qw 4MB | Kw 4MB | Vbf 16MB | O 32MB | ssq 256B  (~56 MB)
  char* ws = (char*)d_ws;
  unsigned short* Qw  = (unsigned short*)ws;
  unsigned short* Kw  = (unsigned short*)(ws + (4u << 20));
  unsigned int*   Vbf = (unsigned int*)(ws + (8u << 20));
  float* O   = (float*)(ws + (24u << 20));
  float* ssq = (float*)(ws + (56u << 20));

  pack_v_kernel<<<16384, 256, 0, stream>>>(f2, tp2, Vbf);
  proj_qk_kernel<<<512, 256, 0, stream>>>(f1, f2, tp1, tp2, W, bias, Qw, Kw, ssq);

  size_t shm = (size_t)NWAVE * 16 * SPITCH * sizeof(float) +
               (size_t)NWAVE * 16 * sizeof(float);
  attn_kernel<<<NBATCH * 16, NWAVE * 32, shm, stream>>>(Qw, Kw, Vbf, O);

  resid_kernel<<<NBATCH * 128, 256, 0, stream>>>(O, f2, tp2, out, ssq);
  scale_kernel<<<8192, 256, 0, stream>>>(out, ssq);
}
